// SparseLinear_82695300317332
// MI455X (gfx1250) — matrix operand order repaired and checked by reading, not compiled, not run on hardware
//
#include <hip/hip_runtime.h>
#include <cstdint>

typedef __attribute__((ext_vector_type(16))) _Float16 v16h;
typedef __attribute__((ext_vector_type(8)))  float    v8f;

#define IN_DIM   1024
#define OUT_DIM  65536
#define BATCH    32
#define COLS_PER_BLOCK 128
#define KTILE    32
#define CSTRIDE  36   // halves per staged column/row (32 data + 4 pad) = 72 B

__global__ __launch_bounds__(256)
void sparse_linear_wmma(const float* __restrict__ x,
                        const float* __restrict__ mask,
                        const float* __restrict__ w,
                        const float* __restrict__ bias,
                        float* __restrict__ out)
{
    // B stage: 128 columns x (32 K-halves + pad), column-major per column
    __shared__ __align__(16) _Float16 Blds[COLS_PER_BLOCK * CSTRIDE]; // 9216 B
    // X stage: 32 rows x (32 K-halves + pad), row-major
    __shared__ __align__(16) _Float16 Xlds[BATCH * CSTRIDE];          // 2304 B

    const int tid  = threadIdx.x;
    const int lane = tid & 31;
    const int wave = tid >> 5;                  // 0..7
    const int n0   = blockIdx.x * COLS_PER_BLOCK;

    const int nloc16 = lane & 15;               // N within 16-wide tile
    const int hi     = lane >> 4;               // 0: lanes 0-15, 1: lanes 16-31
    const int col    = wave * 16 + nloc16;      // 0..127 (block-local column)

    v8f acc0 = {0.f,0.f,0.f,0.f,0.f,0.f,0.f,0.f};
    v8f acc1 = acc0;

    // producer mapping for the W/mask tile: 128 threads across N, 2 groups across K
    const int pn  = tid & 127;                  // column 0..127
    const int pkh = tid >> 7;                   // 0/1 -> K half (16 rows each)

    const uint64_t* bq = (const uint64_t*)Blds; // 9 qwords per column (72 B)
    const uint64_t* xq = (const uint64_t*)Xlds; // 9 qwords per row
    const int bqbase = col * 9 + hi * 2;        // hi lanes start at K=8 (16 B in)
    const int xlo    = hi * 2;                  // hi lanes start at half-offset 8

    for (int kt = 0; kt < IN_DIM / KTILE; ++kt) {
        const int k0 = kt * KTILE;
        __syncthreads();   // previous iteration's fragment reads done

        // ---- stage masked W tile (32K x 128N) as f16, column-major ----
        {
            const size_t gbase = (size_t)(k0 + pkh * 16) * OUT_DIM + n0 + pn;
            const float* wp = w + gbase;
            const float* mp = mask + gbase;
            uint32_t* bcol = (uint32_t*)&Blds[pn * CSTRIDE + pkh * 16];
            #pragma unroll
            for (int j = 0; j < 16; j += 2) {
                float p0 = __builtin_nontemporal_load(wp + (size_t)j * OUT_DIM) *
                           __builtin_nontemporal_load(mp + (size_t)j * OUT_DIM);
                float p1 = __builtin_nontemporal_load(wp + (size_t)(j + 1) * OUT_DIM) *
                           __builtin_nontemporal_load(mp + (size_t)(j + 1) * OUT_DIM);
                union { _Float16 h[2]; uint32_t u; } pk;
                pk.h[0] = (_Float16)p0;
                pk.h[1] = (_Float16)p1;
                bcol[j >> 1] = pk.u;   // conflict-free: bank = (18*pn + ...) mod 64
            }
        }

        // ---- stage X tile (32M x 32K) as f16, row-major ----
        #pragma unroll
        for (int i = 0; i < 4; ++i) {
            int e = tid + 256 * i;
            int m = e >> 5, k = e & 31;
            Xlds[m * CSTRIDE + k] = (_Float16)x[(size_t)m * IN_DIM + k0 + k];
        }

        __syncthreads();   // tiles visible

        // ---- build fragments per the ISA 16-bit A/B VGPR layouts ----
        union Frag { v16h h; uint64_t q[4]; };
        Frag B, A0, A1;
        // B 32x16: lanes0-15 K=0..7 and K=16..23 of column `col`; lanes16-31 +8
        B.q[0] = bq[bqbase + 0];
        B.q[1] = bq[bqbase + 1];
        B.q[2] = bq[bqbase + 4];
        B.q[3] = bq[bqbase + 5];
        // A 16x32: lanes0-15 K=0..7 (V0-3) and K=16..23 (V4-7); lanes16-31 +8
        const int r0 = (0 + nloc16) * 9 + xlo;
        A0.q[0] = xq[r0 + 0]; A0.q[1] = xq[r0 + 1];
        A0.q[2] = xq[r0 + 4]; A0.q[3] = xq[r0 + 5];
        const int r1 = (16 + nloc16) * 9 + xlo;
        A1.q[0] = xq[r1 + 0]; A1.q[1] = xq[r1 + 1];
        A1.q[2] = xq[r1 + 4]; A1.q[3] = xq[r1 + 5];

        acc0 = __builtin_amdgcn_wmma_f32_16x16x32_f16(
                   false, A0.h, false, B.h, (short)0, acc0, false, false);
        acc1 = __builtin_amdgcn_wmma_f32_16x16x32_f16(
                   false, A1.h, false, B.h, (short)0, acc1, false, false);
    }

    // ---- epilogue: D layout lane<16 -> M=v, lane>=16 -> M=8+v; add bias ----
    const int n  = n0 + col;
    const float bn = bias[n];
    const int mb = hi * 8;
    #pragma unroll
    for (int v = 0; v < 8; ++v) {
        __builtin_nontemporal_store(acc0[v] + bn,
            out + (size_t)(mb + v) * OUT_DIM + n);
        __builtin_nontemporal_store(acc1[v] + bn,
            out + (size_t)(16 + mb + v) * OUT_DIM + n);
    }
}

extern "C" void kernel_launch(void* const* d_in, const int* in_sizes, int n_in,
                              void* d_out, int out_size, void* d_ws, size_t ws_size,
                              hipStream_t stream) {
    const float* x    = (const float*)d_in[0];
    const float* mask = (const float*)d_in[1];
    const float* w    = (const float*)d_in[2];
    const float* b    = (const float*)d_in[3];
    float* out = (float*)d_out;

    dim3 grid(OUT_DIM / COLS_PER_BLOCK);   // 512 blocks
    dim3 block(256);                       // 8 waves (wave32)
    sparse_linear_wmma<<<grid, block, 0, stream>>>(x, mask, w, b, out);
}
